// PlayerInteractionGNN_68436008894655
// MI455X (gfx1250) — matrix operand order repaired and checked by reading, not compiled, not run on hardware
//
#include <hip/hip_runtime.h>
#include <hip/hip_bf16.h>
#include <math.h>

// ---------------------------------------------------------------------------
// PlayerInteractionGNN (GATv2, 2 layers, H=4, HID=64) for MI455X / gfx1250.
// Graphs are 10-node fully-connected + self loops -> dense per-graph attention.
// GEMMs: v_wmma_f32_16x16x32_f16, B weights pre-packed into fragment order so
// every lane's B fragment is one aligned 32-byte contiguous load (2x b128).
// Each wave computes a 16x64 C strip: 1 A fragment -> 4 wmma per K-step.
// ---------------------------------------------------------------------------

#define GG   4096          // graphs
#define PP   10            // players per graph
#define NN   (GG * PP)     // nodes = 40960
#define NF   16
#define HID  64
#define OUTD 128
#define NHEAD 4
#define FDIM (NHEAD * HID) // 256
#define NEG_SLOPE 0.2f
#define LN_EPS 1e-5f
#define NT   4             // 16x16 column tiles per wave

typedef __attribute__((ext_vector_type(16))) _Float16 v16h;
typedef __attribute__((ext_vector_type(8)))  _Float16 v8h;
typedef __attribute__((ext_vector_type(8)))  float    v8f;

__device__ __forceinline__ float wave_sum(float v) {
#pragma unroll
  for (int off = 16; off >= 1; off >>= 1) v += __shfl_xor(v, off, 32);
  return v;
}

// ---------------------------------------------------------------------------
// Pack B[K,Nc] (f32 row-major) into WMMA fragment order (f16):
//   Bp[(((nt*ktiles + kt)*32 + lane)*16 + i]
//     = B[kt*32 + (lane>=16 ? 16:0) + i , nt*16 + (lane&15)]
// One fragment = 32 lanes x 16 halves = 1KB contiguous.
// ---------------------------------------------------------------------------
__global__ void k_pack_b(const float* __restrict__ src,
                         _Float16* __restrict__ dst, int K, int Nc) {
  int idx = blockIdx.x * blockDim.x + threadIdx.x;
  if (idx >= K * Nc) return;
  const int i    = idx & 15;
  const int l    = (idx >> 4) & 31;
  const int tile = idx >> 9;          // nt*ktiles + kt
  const int ktiles = K >> 5;
  const int nt = tile / ktiles;
  const int kt = tile - nt * ktiles;
  const int col = (nt << 4) + (l & 15);
  const int kk  = (kt << 5) + ((l >> 4) << 4) + i;
  dst[idx] = (_Float16)src[(size_t)kk * Nc + col];
}

// ---------------------------------------------------------------------------
// Encoder: h0 = relu(LN(x @ W_in + b)) ; x:[N,16] W:[16,64] -> h0 f16 [N,64]
// ---------------------------------------------------------------------------
__global__ void k_encoder(const float* __restrict__ x,
                          const float* __restrict__ W,
                          const float* __restrict__ b,
                          const float* __restrict__ lng,
                          const float* __restrict__ lnb,
                          _Float16* __restrict__ h0) {
  __shared__ float sW[NF * HID];
  __shared__ float sb[HID], sg[HID], sbt[HID];
  for (int i = threadIdx.x; i < NF * HID; i += blockDim.x) sW[i] = W[i];
  for (int i = threadIdx.x; i < HID; i += blockDim.x) {
    sb[i] = b[i]; sg[i] = lng[i]; sbt[i] = lnb[i];
  }
  __syncthreads();

  int node = blockIdx.x * blockDim.x + threadIdx.x;
  if (node >= NN) return;

  float xv[NF];
#pragma unroll
  for (int k = 0; k < NF; ++k) xv[k] = x[(size_t)node * NF + k];

  float acc[HID];
#pragma unroll
  for (int c = 0; c < HID; ++c) acc[c] = sb[c];
  for (int k = 0; k < NF; ++k) {
    float xk = xv[k];
#pragma unroll
    for (int c = 0; c < HID; ++c) acc[c] += xk * sW[k * HID + c];
  }
  float m = 0.f;
#pragma unroll
  for (int c = 0; c < HID; ++c) m += acc[c];
  m *= (1.f / HID);
  float v = 0.f;
#pragma unroll
  for (int c = 0; c < HID; ++c) { float d = acc[c] - m; v += d * d; }
  v *= (1.f / HID);
  float inv = rsqrtf(v + LN_EPS);
#pragma unroll
  for (int c = 0; c < HID; ++c) {
    float y = (acc[c] - m) * inv * sg[c] + sbt[c];
    y = fmaxf(y, 0.f);
    h0[(size_t)node * HID + c] = (_Float16)y;
  }
}

// ---------------------------------------------------------------------------
// WMMA GEMM: C[M,Nc] = A[M,K](f16 row-major) * B(packed fragments) + bias.
// Wave computes 16x(16*NT) strip: A fragment reused NT times per K-step.
// A loads: 2x 16B contiguous (b128). B loads: 1x 32B contiguous per tile.
// ---------------------------------------------------------------------------
__global__ void k_wmma_gemm(const _Float16* __restrict__ A,
                            const _Float16* __restrict__ Bp,
                            const float* __restrict__ bias,
                            float* __restrict__ Cf32,
                            _Float16* __restrict__ Cf16,
                            int M, int Nc, int K, int act_relu) {
  const int lane = threadIdx.x & 31;
  const int wid  = blockIdx.x * (blockDim.x >> 5) + (threadIdx.x >> 5);
  const int ntg  = Nc >> 6;                    // groups of NT=4 col tiles
  const int total = (M >> 4) * ntg;
  if (wid >= total) return;
  const int tm = wid / ntg;
  const int tg = wid - tm * ntg;
  const int ktiles = K >> 5;

  const int arow = (tm << 4) + (lane & 15);    // A: lane holds row M=lane%16
  const int akb  = (lane >> 4) << 4;           // K sub-base 0 / 16 by lane half

  v8f acc[NT];
#pragma unroll
  for (int t = 0; t < NT; ++t) acc[t] = (v8f){};

  for (int kt = 0; kt < ktiles; ++kt) {
    const _Float16* ap = A + (size_t)arow * K + (kt << 5) + akb;
    v8h a0 = *(const v8h*)(ap);                // K = akb + 0..7
    v8h a1 = *(const v8h*)(ap + 8);            // K = akb + 8..15
    v16h a;
#pragma unroll
    for (int i = 0; i < 8; ++i) { a[i] = a0[i]; a[8 + i] = a1[i]; }

#pragma unroll
    for (int t = 0; t < NT; ++t) {
      const int tn = tg * NT + t;
      const _Float16* bpf =
          Bp + ((((size_t)tn * ktiles + kt) << 5) + lane) * 16;
      v16h b = *(const v16h*)bpf;              // 32B aligned contiguous
      acc[t] = __builtin_amdgcn_wmma_f32_16x16x32_f16(
          false, a, false, b, (short)0, acc[t], false, false);
    }
  }

  const int crow0 = (tm << 4) + ((lane >> 4) << 3);
#pragma unroll
  for (int t = 0; t < NT; ++t) {
    const int ccol = ((tg * NT + t) << 4) + (lane & 15);
    const float bv = bias ? bias[ccol] : 0.f;
#pragma unroll
    for (int r = 0; r < 8; ++r) {
      float y = acc[t][r] + bv;
      if (act_relu) y = fmaxf(y, 0.f);
      size_t idx = (size_t)(crow0 + r) * Nc + ccol;
      if (Cf32) Cf32[idx] = y;
      if (Cf16) Cf16[idx] = (_Float16)y;
    }
  }
}

// ---------------------------------------------------------------------------
// GATv2 dense per-graph attention + aggregate + bias + LN + ELU.
// One block (320 threads = 10 waves) per graph; wave i handles dst node i.
// ---------------------------------------------------------------------------
__global__ void k_gat_agg(const float* __restrict__ xl,
                          const float* __restrict__ xr,
                          const float* __restrict__ att,   // [H,HID]
                          const float* __restrict__ bias,
                          const float* __restrict__ lng,
                          const float* __restrict__ lnb,
                          _Float16* __restrict__ out_f16,
                          float* __restrict__ out_f32,
                          int concat) {
  __shared__ float s_xl[PP * FDIM];
  __shared__ float s_xr[PP * FDIM];
  __shared__ float s_att[FDIM];
  __shared__ float s_out[PP * FDIM];

  const int g    = blockIdx.x;
  const int tid  = threadIdx.x;
  const size_t base = (size_t)g * PP * FDIM;
  for (int i = tid; i < PP * FDIM; i += blockDim.x) {
    s_xl[i] = xl[base + i];
    s_xr[i] = xr[base + i];
  }
  for (int i = tid; i < FDIM; i += blockDim.x) s_att[i] = att[i];
  __syncthreads();

  const int i    = tid >> 5;     // dst node 0..9
  const int lane = tid & 31;

  for (int hd = 0; hd < NHEAD; ++hd) {
    const int hb = hd * HID;
    float sc[PP];
#pragma unroll
    for (int j = 0; j < PP; ++j) {
      float p = 0.f;
#pragma unroll
      for (int cc = 0; cc < 2; ++cc) {
        int c = cc * 32 + lane;
        float e = s_xl[j * FDIM + hb + c] + s_xr[i * FDIM + hb + c];
        e = (e > 0.f) ? e : e * NEG_SLOPE;      // leaky_relu
        p += e * s_att[hb + c];
      }
      sc[j] = wave_sum(p);
    }
    float mx = sc[0];
#pragma unroll
    for (int j = 1; j < PP; ++j) mx = fmaxf(mx, sc[j]);
    float al[PP]; float den = 0.f;
#pragma unroll
    for (int j = 0; j < PP; ++j) { al[j] = __expf(sc[j] - mx); den += al[j]; }
    float rd = 1.f / den;
#pragma unroll
    for (int cc = 0; cc < 2; ++cc) {
      int c = cc * 32 + lane;
      float a = 0.f;
#pragma unroll
      for (int j = 0; j < PP; ++j) a += (al[j] * rd) * s_xl[j * FDIM + hb + c];
      s_out[i * FDIM + hb + c] = a;             // same-wave RAW: in-order LDS
    }
  }

  const size_t node = (size_t)g * PP + i;
  if (concat) {
    float vals[8]; float s = 0.f;
#pragma unroll
    for (int r = 0; r < 8; ++r) {
      int c = r * 32 + lane;
      float v = s_out[i * FDIM + c] + bias[c];
      vals[r] = v; s += v;
    }
    s = wave_sum(s) * (1.f / FDIM);
    float q = 0.f;
#pragma unroll
    for (int r = 0; r < 8; ++r) { float d = vals[r] - s; q += d * d; }
    q = wave_sum(q) * (1.f / FDIM);
    float inv = rsqrtf(q + LN_EPS);
#pragma unroll
    for (int r = 0; r < 8; ++r) {
      int c = r * 32 + lane;
      float y = (vals[r] - s) * inv * lng[c] + lnb[c];
      y = (y > 0.f) ? y : (__expf(y) - 1.f);    // elu
      out_f16[node * FDIM + c] = (_Float16)y;
    }
  } else {
    float vals[2]; float s = 0.f;
#pragma unroll
    for (int r = 0; r < 2; ++r) {
      int c = r * 32 + lane;
      float v = 0.25f * (s_out[i * FDIM + c] + s_out[i * FDIM + HID + c] +
                         s_out[i * FDIM + 2 * HID + c] + s_out[i * FDIM + 3 * HID + c]);
      v += bias[c];
      vals[r] = v; s += v;
    }
    s = wave_sum(s) * (1.f / HID);
    float q = 0.f;
#pragma unroll
    for (int r = 0; r < 2; ++r) { float d = vals[r] - s; q += d * d; }
    q = wave_sum(q) * (1.f / HID);
    float inv = rsqrtf(q + LN_EPS);
#pragma unroll
    for (int r = 0; r < 2; ++r) {
      int c = r * 32 + lane;
      float y = (vals[r] - s) * inv * lng[c] + lnb[c];
      y = (y > 0.f) ? y : (__expf(y) - 1.f);
      out_f32[node * HID + c] = y;
    }
  }
}

// ---------------------------------------------------------------------------
// Per-graph mean/max pool: h[N,64] -> z16[G,128] (mean || max).
// ---------------------------------------------------------------------------
__global__ void k_pool(const float* __restrict__ h, _Float16* __restrict__ z16) {
  int g = blockIdx.x;
  int c = threadIdx.x;   // 64
  float s = 0.f, mx = -INFINITY;
#pragma unroll
  for (int p = 0; p < PP; ++p) {
    float v = h[((size_t)g * PP + p) * HID + c];
    s += v; mx = fmaxf(mx, v);
  }
  s *= (1.f / PP);
  z16[(size_t)g * 2 * HID + c]       = (_Float16)s;
  z16[(size_t)g * 2 * HID + HID + c] = (_Float16)mx;
}

// ---------------------------------------------------------------------------
// Final LayerNorm over 128 dims: one wave per row.
// ---------------------------------------------------------------------------
__global__ void k_final_ln(const float* __restrict__ y,
                           const float* __restrict__ lng,
                           const float* __restrict__ lnb,
                           float* __restrict__ out) {
  int row  = blockIdx.x * (blockDim.x >> 5) + (threadIdx.x >> 5);
  int lane = threadIdx.x & 31;
  if (row >= GG) return;
  float v[4]; float s = 0.f;
#pragma unroll
  for (int r = 0; r < 4; ++r) {
    v[r] = y[(size_t)row * OUTD + r * 32 + lane];
    s += v[r];
  }
  s = wave_sum(s) * (1.f / OUTD);
  float q = 0.f;
#pragma unroll
  for (int r = 0; r < 4; ++r) { float d = v[r] - s; q += d * d; }
  q = wave_sum(q) * (1.f / OUTD);
  float inv = rsqrtf(q + LN_EPS);
#pragma unroll
  for (int r = 0; r < 4; ++r) {
    int c = r * 32 + lane;
    out[(size_t)row * OUTD + c] = (v[r] - s) * inv * lng[c] + lnb[c];
  }
}

// ---------------------------------------------------------------------------
// Host launch
// ---------------------------------------------------------------------------
static inline size_t align_up(size_t x) { return (x + 255) & ~(size_t)255; }

extern "C" void kernel_launch(void* const* d_in, const int* in_sizes, int n_in,
                              void* d_out, int out_size, void* d_ws, size_t ws_size,
                              hipStream_t stream) {
  (void)in_sizes; (void)n_in; (void)out_size; (void)ws_size;

  const float* x      = (const float*)d_in[0];
  // d_in[1] = edge_index, d_in[2] = batch: unused (static all-to-all graphs).
  const float* in_W   = (const float*)d_in[3];
  const float* in_b   = (const float*)d_in[4];
  const float* in_lg  = (const float*)d_in[5];
  const float* in_lb  = (const float*)d_in[6];
  const float* op_W1  = (const float*)d_in[7];
  const float* op_b1  = (const float*)d_in[8];
  const float* op_W2  = (const float*)d_in[9];
  const float* op_b2  = (const float*)d_in[10];
  const float* o_lg   = (const float*)d_in[11];
  const float* o_lb   = (const float*)d_in[12];
  const float* Wl0 = (const float*)d_in[13];
  const float* bl0 = (const float*)d_in[14];
  const float* Wr0 = (const float*)d_in[15];
  const float* br0 = (const float*)d_in[16];
  const float* at0 = (const float*)d_in[17];
  const float* bi0 = (const float*)d_in[18];
  const float* lg0 = (const float*)d_in[19];
  const float* lb0 = (const float*)d_in[20];
  const float* Wl1 = (const float*)d_in[21];
  const float* bl1 = (const float*)d_in[22];
  const float* Wr1 = (const float*)d_in[23];
  const float* br1 = (const float*)d_in[24];
  const float* at1 = (const float*)d_in[25];
  const float* bi1 = (const float*)d_in[26];
  const float* lg1 = (const float*)d_in[27];
  const float* lb1 = (const float*)d_in[28];

  float* out = (float*)d_out;

  // ---- workspace bump allocator ----
  char* w = (char*)d_ws;
  auto alloc = [&](size_t bytes) -> void* {
    void* p = (void*)w; w += align_up(bytes); return p;
  };
  float*     xlb   = (float*)alloc((size_t)NN * FDIM * 4);
  float*     xrb   = (float*)alloc((size_t)NN * FDIM * 4);
  _Float16*  hf16  = (_Float16*)alloc((size_t)NN * FDIM * 2);
  float*     hfin  = (float*)alloc((size_t)NN * HID * 4);
  _Float16*  z16   = (_Float16*)alloc((size_t)GG * 2 * HID * 2);
  float*     m32   = (float*)alloc((size_t)GG * HID * 4);
  _Float16*  m16   = (_Float16*)alloc((size_t)GG * HID * 2);
  float*     y32   = (float*)alloc((size_t)GG * OUTD * 4);
  _Float16*  Wl0p  = (_Float16*)alloc((size_t)HID * FDIM * 2);
  _Float16*  Wr0p  = (_Float16*)alloc((size_t)HID * FDIM * 2);
  _Float16*  Wl1p  = (_Float16*)alloc((size_t)FDIM * FDIM * 2);
  _Float16*  Wr1p  = (_Float16*)alloc((size_t)FDIM * FDIM * 2);
  _Float16*  W1p   = (_Float16*)alloc((size_t)(2 * HID) * HID * 2);
  _Float16*  W2p   = (_Float16*)alloc((size_t)HID * OUTD * 2);

  auto pack = [&](const float* s, _Float16* d, int K, int Nc) {
    int n = K * Nc;
    k_pack_b<<<(n + 255) / 256, 256, 0, stream>>>(s, d, K, Nc);
  };
  pack(Wl0, Wl0p, HID, FDIM);
  pack(Wr0, Wr0p, HID, FDIM);
  pack(Wl1, Wl1p, FDIM, FDIM);
  pack(Wr1, Wr1p, FDIM, FDIM);
  pack(op_W1, W1p, 2 * HID, HID);
  pack(op_W2, W2p, HID, OUTD);

  auto gemm = [&](const _Float16* A, const _Float16* Bp, const float* bias,
                  float* C32, _Float16* C16, int M, int Nc, int K, int relu) {
    int waves = (M / 16) * (Nc / (16 * NT));
    int blocks = (waves + 7) / 8;                 // 8 waves / block (256 thr)
    k_wmma_gemm<<<blocks, 256, 0, stream>>>(A, Bp, bias, C32, C16, M, Nc, K, relu);
  };

  // 1) Encoder: h0 = relu(LN(x @ W_in + b))  -> hf16 [N,64]
  k_encoder<<<(NN + 127) / 128, 128, 0, stream>>>(x, in_W, in_b, in_lg, in_lb, hf16);

  // 2) Layer 0 projections: xl0/xr0 = h0 @ Wl0/Wr0 + b  [N,256]
  gemm(hf16, Wl0p, bl0, xlb, nullptr, NN, FDIM, HID, 0);
  gemm(hf16, Wr0p, br0, xrb, nullptr, NN, FDIM, HID, 0);

  // 3) Layer 0 attention + aggregate + LN + ELU -> h1 f16 [N,256]
  k_gat_agg<<<GG, 320, 0, stream>>>(xlb, xrb, at0, bi0, lg0, lb0,
                                    hf16, nullptr, /*concat=*/1);

  // 4) Layer 1 projections: xl1/xr1 = h1 @ Wl1/Wr1 + b  [N,256]
  gemm(hf16, Wl1p, bl1, xlb, nullptr, NN, FDIM, FDIM, 0);
  gemm(hf16, Wr1p, br1, xrb, nullptr, NN, FDIM, FDIM, 0);

  // 5) Layer 1 attention + head-mean + LN + ELU -> h_final f32 [N,64]
  k_gat_agg<<<GG, 320, 0, stream>>>(xlb, xrb, at1, bi1, lg1, lb1,
                                    nullptr, hfin, /*concat=*/0);

  // 6) Per-graph mean||max pool -> z16 [G,128]
  k_pool<<<GG, HID, 0, stream>>>(hfin, z16);

  // 7) MLP: relu(z @ W1 + b1) -> m  [G,64] ; m @ W2 + b2 -> y [G,128]
  gemm(z16, W1p, op_b1, m32, m16, GG, HID, 2 * HID, 1);
  gemm(m16, W2p, op_b2, y32, nullptr, GG, OUTD, HID, 0);

  // 8) Final LayerNorm -> d_out
  k_final_ln<<<GG / 8, 256, 0, stream>>>(y32, o_lg, o_lb, out);
}
